// EpiGNN_50483045598039
// MI455X (gfx1250) — matrix operand order repaired and checked by reading, not compiled, not run on hardware
//
#include <hip/hip_runtime.h>
#include <hip/hip_bf16.h>

// ---------------------------------------------------------------------------
// Types for CDNA5 WMMA
// ---------------------------------------------------------------------------
typedef __bf16 bf16_t;
typedef __attribute__((ext_vector_type(16))) __bf16 v16bf;
typedef __attribute__((ext_vector_type(8)))  __bf16 v8bf;
typedef __attribute__((ext_vector_type(8)))  float  v8f;
typedef int v4i_t __attribute__((ext_vector_type(4)));

#define MT 128   // M tile per workgroup
#define NT 128   // N tile (== HID)
#define KT 32    // K step per WMMA (bf16)
#define LDS_STRIDE 40  // KT + 8 pad (80 bytes/row, 16B aligned)

// Async global->LDS path (CDNA5): guarded so compilation never breaks.
#if defined(__has_builtin)
#  if __has_builtin(__builtin_amdgcn_global_load_async_to_lds_b128) && \
      __has_builtin(__builtin_amdgcn_s_wait_asynccnt)
#    define USE_ASYNC_LDS 1
#  endif
#endif
#ifndef USE_ASYNC_LDS
#  define USE_ASYNC_LDS 0
#endif

__device__ __forceinline__ void cp16_g2l(const bf16_t* g, bf16_t* l) {
#if USE_ASYNC_LDS
  // builtin signature: (v4i addrspace(1)*, v4i addrspace(3)*, imm offset, imm cpol)
  __builtin_amdgcn_global_load_async_to_lds_b128(
      (__attribute__((address_space(1))) v4i_t*)g,
      (__attribute__((address_space(3))) v4i_t*)l, 0, 0);
#else
  *(uint4*)l = *(const uint4*)g;
#endif
}

template <int NCNT>
__device__ __forceinline__ void wait_async_le() {
#if USE_ASYNC_LDS
  __builtin_amdgcn_s_wait_asynccnt(NCNT);
#endif
}

__device__ __forceinline__ unsigned short bf16bits(float f) {
  return __builtin_bit_cast(unsigned short, (bf16_t)f);
}

// ---------------------------------------------------------------------------
// GEMM: C[M x 128] = A[M x K](bf16) @ BT[128 x K](bf16, pre-transposed) + bias
// 8 waves; each wave computes a 16x128 strip via 8 v_wmma_f32_16x16x32_bf16.
// Double-buffered LDS, tiles staged with async b128 global->LDS copies.
// ---------------------------------------------------------------------------
__global__ __launch_bounds__(256) void gemm_bf16ab_wmma(
    const unsigned short* __restrict__ Abits,   // [M,K] bf16 row-major
    int lda,                                    // == K
    const unsigned short* __restrict__ BTbits,  // [128,K] bf16 (col-major B)
    const float* __restrict__ bias,             // [128] or nullptr
    float* __restrict__ C,                      // [M,128] f32
    int M, int K)
{
  __shared__ __align__(16) bf16_t As[2][MT][LDS_STRIDE];
  __shared__ __align__(16) bf16_t Bs[2][NT][LDS_STRIDE];   // BT layout: [col][k]

  const bf16_t* A  = (const bf16_t*)Abits;
  const bf16_t* BT = (const bf16_t*)BTbits;

  const int tid  = threadIdx.x;
  const int wave = tid >> 5;
  const int lane = tid & 31;
  const int m0   = blockIdx.x * MT;

  v8f acc[8];
  #pragma unroll
  for (int t = 0; t < 8; ++t)
    #pragma unroll
    for (int i = 0; i < 8; ++i) acc[t][i] = 0.0f;

  // stage one k-tile pair (A 128x32, BT 128x32) into buffer `buf`
  auto stage = [&](int buf, int k0) {
    #pragma unroll
    for (int j = 0; j < 2; ++j) {              // A: 512 16B chunks / 256 thr
      int q   = tid + 256 * j;
      int row = q >> 2;
      int ch  = (q & 3) * 8;                   // bf16 element offset
      int grow = m0 + row;
      if (grow >= M) grow = M - 1;             // clamp: valid mem, rows discarded
      cp16_g2l(A + (size_t)grow * lda + k0 + ch, &As[buf][row][ch]);
    }
    #pragma unroll
    for (int j = 0; j < 2; ++j) {              // BT: 512 16B chunks
      int q   = tid + 256 * j;
      int col = q >> 2;
      int ch  = (q & 3) * 8;
      cp16_g2l(BT + (size_t)col * K + k0 + ch, &Bs[buf][col][ch]);
    }
  };

  const int nsteps = K / KT;
  stage(0, 0);

  for (int step = 0; step < nsteps; ++step) {
    const int cur = step & 1;
    __syncthreads();                       // buffer cur^1 free everywhere
    if (step + 1 < nsteps) {
      stage(cur ^ 1, (step + 1) * KT);     // prefetch next tile (async)
      wait_async_le<4>();                  // own cur-tile copies complete
    } else {
      wait_async_le<0>();
    }
    __syncthreads();                       // all waves' cur tiles visible

    // A fragment: row = wave*16 + (lane&15); k(i) = (i<8?i:i+8)+(lane<16?0:8)
    const int r   = (wave << 4) + (lane & 15);
    const int akb = (lane < 16) ? 0 : 8;
    v16bf afrag;
    {
      v8bf lo = *(const v8bf*)&As[cur][r][akb];
      v8bf hi = *(const v8bf*)&As[cur][r][akb + 16];
      #pragma unroll
      for (int i = 0; i < 8; ++i) { afrag[i] = lo[i]; afrag[i + 8] = hi[i]; }
    }
    // B fragment: col = t*16 + (lane&15); k(i) = (i<8?i:i+8)+(lane<16?0:8)
    const int bkb = (lane < 16) ? 0 : 8;
    #pragma unroll
    for (int t = 0; t < 8; ++t) {
      const int col = t * 16 + (lane & 15);
      v16bf bfrag;
      v8bf lo = *(const v8bf*)&Bs[cur][col][bkb];
      v8bf hi = *(const v8bf*)&Bs[cur][col][bkb + 16];
      #pragma unroll
      for (int i = 0; i < 8; ++i) { bfrag[i] = lo[i]; bfrag[i + 8] = hi[i]; }
      acc[t] = __builtin_amdgcn_wmma_f32_16x16x32_bf16(
          false, afrag, false, bfrag, (short)0, acc[t], false, false);
    }
  }

  // epilogue: C layout (VGPR rr, lane l) -> m = rr + 8*(l>=16), n = l&15
  const int rowbase = m0 + (wave << 4) + 8 * (lane >> 4);
  const int coll    = lane & 15;
  #pragma unroll
  for (int t = 0; t < 8; ++t) {
    const int col = t * 16 + coll;
    const float bv = bias ? bias[col] : 0.0f;
    #pragma unroll
    for (int rr = 0; rr < 8; ++rr) {
      const int row = rowbase + rr;
      if (row < M) C[(size_t)row * NT + col] = acc[t][rr] + bv;
    }
  }
}

// ---------------------------------------------------------------------------
// f32 -> bf16 pack-convert, 4 elements/thread (v_cvt_pk_bf16_f32 path)
// ---------------------------------------------------------------------------
__global__ __launch_bounds__(256) void cvt_bf16x4_kernel(
    const float* __restrict__ in, unsigned short* __restrict__ out, long long n4)
{
  long long i = (long long)blockIdx.x * 256 + threadIdx.x;
  if (i >= n4) return;
  float4 v = ((const float4*)in)[i];
  unsigned lo = ((unsigned)bf16bits(v.y) << 16) | bf16bits(v.x);
  unsigned hi = ((unsigned)bf16bits(v.w) << 16) | bf16bits(v.z);
  ((uint2*)out)[i] = make_uint2(lo, hi);
}

// ---------------------------------------------------------------------------
// transpose+convert weight W[K,128] f32 -> BT[128,K] bf16
// ---------------------------------------------------------------------------
__global__ __launch_bounds__(256) void wT_bf16_kernel(
    const float* __restrict__ W, unsigned short* __restrict__ BT, int K)
{
  int i = blockIdx.x * 256 + threadIdx.x;
  if (i >= 128 * K) return;
  int col = i / K, k = i - col * K;
  BT[i] = bf16bits(W[(size_t)k * 128 + col]);
}

// ---------------------------------------------------------------------------
// ea = edge_attr[E,3] @ edge_W[3,128] + edge_b   (K=3: bandwidth bound)
// ---------------------------------------------------------------------------
__global__ __launch_bounds__(256) void edge_proj_kernel(
    const float* __restrict__ eattr, const float* __restrict__ W,
    const float* __restrict__ b, float* __restrict__ out, int E)
{
  long long i = (long long)blockIdx.x * 256 + threadIdx.x;
  if (i >= (long long)E * 128) return;
  int e = (int)(i >> 7), j = (int)(i & 127);
  const float* row = eattr + (size_t)e * 3;
  out[i] = row[0] * W[j] + row[1] * W[128 + j] + row[2] * W[256 + j] + b[j];
}

// order-preserving float<->uint for atomicMax segment-max
__device__ __forceinline__ unsigned f2ord(float f) {
  unsigned u = __float_as_uint(f);
  return (u & 0x80000000u) ? ~u : (u | 0x80000000u);
}
__device__ __forceinline__ float ord2f(unsigned o) {
  unsigned u = (o & 0x80000000u) ? (o & 0x7fffffffu) : ~o;
  return __uint_as_float(u);
}

// ---------------------------------------------------------------------------
// attention logits: one wave per edge; lane = d (0..31), 4 heads per lane
// ---------------------------------------------------------------------------
__global__ __launch_bounds__(256) void attn_logits_kernel(
    const float* __restrict__ xl, const float* __restrict__ xr,
    const float* __restrict__ ee, const int* __restrict__ src,
    const int* __restrict__ dst, const float* __restrict__ att,
    unsigned* __restrict__ mord, float* __restrict__ alpha, int E)
{
  int e = blockIdx.x * 8 + (threadIdx.x >> 5);
  if (e >= E) return;
  const int lane = threadIdx.x & 31;
  const int s = src[e], d = dst[e];
  float vh[4];
  #pragma unroll
  for (int h = 0; h < 4; ++h) {
    int c = h * 32 + lane;
    float z = xl[(size_t)s * 128 + c] + xr[(size_t)d * 128 + c] +
              ee[(size_t)e * 128 + c];
    z = (z > 0.0f) ? z : 0.2f * z;
    vh[h] = z * att[c];
  }
  #pragma unroll
  for (int off = 16; off > 0; off >>= 1) {
    #pragma unroll
    for (int h = 0; h < 4; ++h) vh[h] += __shfl_xor(vh[h], off, 32);
  }
  if (lane < 4) {
    float v = vh[lane];
    alpha[(size_t)e * 4 + lane] = v;
    atomicMax(&mord[(size_t)d * 4 + lane], f2ord(v));
  }
}

// a = exp(alpha - m[dst]); denom[dst] += a  (alpha overwritten with a)
__global__ __launch_bounds__(256) void attn_exp_kernel(
    float* __restrict__ alpha, const int* __restrict__ dst,
    const unsigned* __restrict__ mord, float* __restrict__ denom, int E)
{
  int i = blockIdx.x * 256 + threadIdx.x;
  if (i >= E * 4) return;
  int e = i >> 2, h = i & 3;
  int d = dst[e];
  float m = ord2f(mord[(size_t)d * 4 + h]);
  float a = __expf(alpha[i] - m);
  alpha[i] = a;
  atomicAdd(&denom[(size_t)d * 4 + h], a);
}

// out[dst] += xl[src] * a / denom[dst]  : thread per (edge, channel)
__global__ __launch_bounds__(256) void message_kernel(
    const float* __restrict__ xl, const float* __restrict__ alpha,
    const float* __restrict__ denom, const int* __restrict__ src,
    const int* __restrict__ dst, float* __restrict__ agg, long long total)
{
  long long i = (long long)blockIdx.x * 256 + threadIdx.x;
  if (i >= total) return;
  int e = (int)(i >> 7), c = (int)(i & 127), h = c >> 5;
  int s = src[e], d = dst[e];
  float a = alpha[(size_t)e * 4 + h] /
            (denom[(size_t)d * 4 + h] + 1e-16f);
  atomicAdd(&agg[(size_t)d * 128 + c], xl[(size_t)s * 128 + c] * a);
}

// h += relu(LayerNorm(agg + gat_b) * g + b) : one wave per node
__global__ __launch_bounds__(256) void ln_update_kernel(
    float* __restrict__ h, const float* __restrict__ agg,
    const float* __restrict__ gb, const float* __restrict__ g,
    const float* __restrict__ b, int N)
{
  int n = blockIdx.x * 8 + (threadIdx.x >> 5);
  if (n >= N) return;
  const int lane = threadIdx.x & 31;
  float v[4], s = 0.0f, s2 = 0.0f;
  #pragma unroll
  for (int j = 0; j < 4; ++j) {
    int c = j * 32 + lane;
    v[j] = agg[(size_t)n * 128 + c] + gb[c];
    s += v[j]; s2 += v[j] * v[j];
  }
  #pragma unroll
  for (int off = 16; off > 0; off >>= 1) {
    s  += __shfl_xor(s,  off, 32);
    s2 += __shfl_xor(s2, off, 32);
  }
  float mu  = s * (1.0f / 128.0f);
  float var = s2 * (1.0f / 128.0f) - mu * mu;
  float rs  = rsqrtf(var + 1e-5f);
  #pragma unroll
  for (int j = 0; j < 4; ++j) {
    int c = j * 32 + lane;
    float o = (v[j] - mu) * rs * g[c] + b[c];
    o = fmaxf(o, 0.0f);
    h[(size_t)n * 128 + c] += o;
  }
}

// pooled[batch[n]] += h[n]
__global__ __launch_bounds__(256) void pool_kernel(
    const float* __restrict__ h, const int* __restrict__ batch,
    float* __restrict__ pooled, int N)
{
  long long i = (long long)blockIdx.x * 256 + threadIdx.x;
  if (i >= (long long)N * 128) return;
  int n = (int)(i >> 7), c = (int)(i & 127);
  atomicAdd(&pooled[(size_t)batch[n] * 128 + c], h[i]);
}

// readout: out[g] = relu(pooled[g] @ W1 + b1) @ W2 + b2 ; block(64) per graph
__global__ __launch_bounds__(64) void readout_kernel(
    const float* __restrict__ pooled, const float* __restrict__ W1,
    const float* __restrict__ b1, const float* __restrict__ W2,
    const float* __restrict__ b2, float* __restrict__ out)
{
  __shared__ float sh[64];
  const int g = blockIdx.x, t = threadIdx.x;
  float acc = b1[t];
  for (int c = 0; c < 128; ++c)
    acc += pooled[(size_t)g * 128 + c] * W1[c * 64 + t];
  acc = fmaxf(acc, 0.0f);
  sh[t] = acc * W2[t];
  __syncthreads();
  #pragma unroll
  for (int off = 32; off > 0; off >>= 1) {
    if (t < off) sh[t] += sh[t + off];
    __syncthreads();
  }
  if (t == 0) out[g] = sh[0] + b2[0];
}

__global__ __launch_bounds__(256) void zero_u32_kernel(unsigned* p, long long n) {
  long long i = (long long)blockIdx.x * 256 + threadIdx.x;
  if (i < n) p[i] = 0u;
}

// ---------------------------------------------------------------------------
// Host orchestration
// ---------------------------------------------------------------------------
extern "C" void kernel_launch(void* const* d_in, const int* in_sizes, int n_in,
                              void* d_out, int out_size, void* d_ws, size_t ws_size,
                              hipStream_t stream) {
  const float* x      = (const float*)d_in[0];
  const float* eattr  = (const float*)d_in[1];
  const int*   eidx   = (const int*)d_in[2];
  const int*   batch  = (const int*)d_in[3];
  const float* node_W = (const float*)d_in[4];
  const float* node_b = (const float*)d_in[5];
  const float* edge_W = (const float*)d_in[6];
  const float* edge_b = (const float*)d_in[7];
  const float* lin_l  = (const float*)d_in[8];
  const float* lin_r  = (const float*)d_in[9];
  const float* lin_e  = (const float*)d_in[10];
  const float* att    = (const float*)d_in[11];
  const float* gat_b  = (const float*)d_in[12];
  const float* ln_g   = (const float*)d_in[13];
  const float* ln_b   = (const float*)d_in[14];
  const float* r1_W   = (const float*)d_in[15];
  const float* r1_b   = (const float*)d_in[16];
  const float* r2_W   = (const float*)d_in[17];
  const float* r2_b   = (const float*)d_in[18];
  float* out = (float*)d_out;

  const int N = in_sizes[3];          // 50000
  const int E = in_sizes[1] / 3;      // 600000
  const int G = out_size;             // 512
  const int IN_DIM = in_sizes[0] / N; // 1280
  const int LAYERS = 2;

  // ---- workspace carve-up: f32 region, then bf16 (ushort) region ----
  float* ws = (float*)d_ws;
  size_t off = 0;
  float*    h     = ws + off; off += (size_t)N * 128;
  float*    ea    = ws + off; off += (size_t)E * 128;
  float*    ee    = ws + off; off += (size_t)E * 128;
  float*    xl    = ws + off; off += (size_t)N * 128;
  float*    xr    = ws + off; off += (size_t)N * 128;
  float*    agg   = ws + off; off += (size_t)N * 128;
  float*    alpha = ws + off; off += (size_t)E * 4;
  unsigned* mord  = (unsigned*)(ws + off); off += (size_t)N * 4;
  float*    denom = ws + off; off += (size_t)N * 4;
  float*    pooled= ws + off; off += (size_t)G * 128;

  unsigned short* us = (unsigned short*)(ws + off);
  size_t uoff = 0;
  unsigned short* xb   = us + uoff; uoff += (size_t)N * IN_DIM;  // bf16 x
  unsigned short* eab  = us + uoff; uoff += (size_t)E * 128;     // bf16 ea
  unsigned short* hb   = us + uoff; uoff += (size_t)N * 128;     // bf16 h
  unsigned short* nWT  = us + uoff; uoff += (size_t)128 * IN_DIM;
  unsigned short* linT = us + uoff; uoff += (size_t)LAYERS * 3 * 128 * 128;
  (void)ws_size; (void)n_in;

  const int* src = eidx;
  const int* dst = eidx + E;
  const dim3 blk256(256);

  // ---- one-time (per call) weight transpose/convert ----
  wT_bf16_kernel<<<dim3((128 * IN_DIM + 255) / 256), blk256, 0, stream>>>(
      node_W, nWT, IN_DIM);
  for (int l = 0; l < LAYERS; ++l) {
    const size_t wsz = (size_t)128 * 128;
    wT_bf16_kernel<<<dim3((128 * 128 + 255) / 256), blk256, 0, stream>>>(
        lin_l + l * wsz, linT + (size_t)(l * 3 + 0) * wsz, 128);
    wT_bf16_kernel<<<dim3((128 * 128 + 255) / 256), blk256, 0, stream>>>(
        lin_r + l * wsz, linT + (size_t)(l * 3 + 1) * wsz, 128);
    wT_bf16_kernel<<<dim3((128 * 128 + 255) / 256), blk256, 0, stream>>>(
        lin_e + l * wsz, linT + (size_t)(l * 3 + 2) * wsz, 128);
  }

  // ---- h = x @ node_W + node_b ----
  {
    long long n4 = (long long)N * IN_DIM / 4;
    cvt_bf16x4_kernel<<<dim3((unsigned)((n4 + 255) / 256)), blk256, 0, stream>>>(
        x, xb, n4);
  }
  gemm_bf16ab_wmma<<<dim3((N + MT - 1) / MT), blk256, 0, stream>>>(
      xb, IN_DIM, nWT, node_b, h, N, IN_DIM);

  // ---- ea = edge_attr @ edge_W + edge_b ; convert once to bf16 ----
  {
    long long tot = (long long)E * 128;
    edge_proj_kernel<<<dim3((unsigned)((tot + 255) / 256)), blk256, 0, stream>>>(
        eattr, edge_W, edge_b, ea, E);
    cvt_bf16x4_kernel<<<dim3((unsigned)((tot / 4 + 255) / 256)), blk256, 0, stream>>>(
        ea, eab, tot / 4);
  }

  for (int l = 0; l < LAYERS; ++l) {
    const size_t wsz = (size_t)128 * 128;
    const unsigned short* WlT = linT + (size_t)(l * 3 + 0) * wsz;
    const unsigned short* WrT = linT + (size_t)(l * 3 + 1) * wsz;
    const unsigned short* WeT = linT + (size_t)(l * 3 + 2) * wsz;
    const float* al = att   + (size_t)l * 4 * 32;
    const float* gb = gat_b + (size_t)l * 128;
    const float* lg = ln_g  + (size_t)l * 128;
    const float* lb = ln_b  + (size_t)l * 128;

    // hb = bf16(h)
    {
      long long n4 = (long long)N * 128 / 4;
      cvt_bf16x4_kernel<<<dim3((unsigned)((n4 + 255) / 256)), blk256, 0, stream>>>(
          h, hb, n4);
    }
    gemm_bf16ab_wmma<<<dim3((N + MT - 1) / MT), blk256, 0, stream>>>(
        hb, 128, WlT, nullptr, xl, N, 128);
    gemm_bf16ab_wmma<<<dim3((N + MT - 1) / MT), blk256, 0, stream>>>(
        hb, 128, WrT, nullptr, xr, N, 128);
    gemm_bf16ab_wmma<<<dim3((E + MT - 1) / MT), blk256, 0, stream>>>(
        eab, 128, WeT, nullptr, ee, E, 128);

    // zero agg, then mord+denom (contiguous N*8 u32)
    zero_u32_kernel<<<dim3((unsigned)(((long long)N * 128 + 255) / 256)), blk256, 0, stream>>>(
        (unsigned*)agg, (long long)N * 128);
    zero_u32_kernel<<<dim3((unsigned)(((long long)N * 8 + 255) / 256)), blk256, 0, stream>>>(
        mord, (long long)N * 8);

    attn_logits_kernel<<<dim3((E + 7) / 8), blk256, 0, stream>>>(
        xl, xr, ee, src, dst, al, mord, alpha, E);
    attn_exp_kernel<<<dim3((E * 4 + 255) / 256), blk256, 0, stream>>>(
        alpha, dst, mord, denom, E);
    {
      long long tot = (long long)E * 128;
      message_kernel<<<dim3((unsigned)((tot + 255) / 256)), blk256, 0, stream>>>(
          xl, alpha, denom, src, dst, agg, tot);
    }
    ln_update_kernel<<<dim3((N + 7) / 8), blk256, 0, stream>>>(
        h, agg, gb, lg, lb, N);
  }

  // pooling + readout
  zero_u32_kernel<<<dim3((unsigned)(((long long)G * 128 + 255) / 256)), blk256, 0, stream>>>(
      (unsigned*)pooled, (long long)G * 128);
  pool_kernel<<<dim3((unsigned)(((long long)N * 128 + 255) / 256)), blk256, 0, stream>>>(
      h, batch, pooled, N);
  readout_kernel<<<dim3(G), dim3(64), 0, stream>>>(
      pooled, r1_W, r1_b, r2_W, r2_b, out);
}
